// MoELayer_89094801588254
// MI455X (gfx1250) — matrix operand order repaired and checked by reading, not compiled, not run on hardware
//
#include <hip/hip_runtime.h>
#include <hip/hip_bf16.h>
#include <cmath>

// ---------------- problem constants ----------------
#define NTOK   32768        // 8 * 4096 tokens
#define DIM    768
#define DFF    3072
#define NEXP   8
#define MT     32           // tokens per ffn tile
#define CHUNK  64           // d_ff chunk per iteration
#define MAXTILES (NTOK / MT + NEXP)   // 1032, worst case
#define OUT_ELEMS (NTOK * DIM)        // 25165824

// fragment counts for pre-swizzled bf16 weights
#define W1_NT   (DFF / 16)            // 192 N-tiles
#define W1_KS   (DIM / 32)            // 24 K-steps
#define W2_NT   (DIM / 16)            // 48 N-tiles
#define W2_KS   (DFF / 32)            // 96 K-steps
#define W1_FRAGS (NEXP * W1_NT * W1_KS)   // 36864
#define W2_FRAGS (NEXP * W2_KS * W2_NT)   // 36864
// each fragment = 32 lanes * 8 packed dwords = 1024 bytes

typedef __attribute__((ext_vector_type(16))) __bf16 bf16x16;
typedef __attribute__((ext_vector_type(8)))  float  f32x8;

union Frag {
    bf16x16 v;
    unsigned int u[8];
};

__device__ __forceinline__ unsigned short f2bf(float f) {
    unsigned int u = __float_as_uint(f);
    u += 0x7FFFu + ((u >> 16) & 1u);          // round-to-nearest-even
    return (unsigned short)(u >> 16);
}

__device__ __forceinline__ void loadFragG(const unsigned int* __restrict__ fp, Frag& B) {
    uint4 b0 = *(const uint4*)(fp);
    uint4 b1 = *(const uint4*)(fp + 4);
    B.u[0] = b0.x; B.u[1] = b0.y; B.u[2] = b0.z; B.u[3] = b0.w;
    B.u[4] = b1.x; B.u[5] = b1.y; B.u[6] = b1.z; B.u[7] = b1.w;
}

// ---------------- 0: zero the tiny counters ----------------
__global__ void init_kernel(int* counts, float* probsum) {
    int t = threadIdx.x;
    if (t < NEXP) { counts[t] = 0; probsum[t] = 0.0f; }
}

// ---------------- 1: gating (one wave32 per token) ----------------
__global__ __launch_bounds__(256) void gate_kernel(
    const float* __restrict__ x, const float* __restrict__ gw,
    int* __restrict__ expert_of, float* __restrict__ w_of,
    int* __restrict__ counts, float* __restrict__ probsum)
{
    __shared__ float sProb[NEXP];
    __shared__ int   sCnt[NEXP];
    const int tid = threadIdx.x;
    if (tid < NEXP) { sProb[tid] = 0.0f; sCnt[tid] = 0; }
    __syncthreads();

    const int w = tid >> 5, lane = tid & 31;
    const int t = blockIdx.x * 8 + w;                 // 8 waves/block
    const float* xr = x + (size_t)t * DIM;

    float acc[NEXP];
#pragma unroll
    for (int e = 0; e < NEXP; ++e) acc[e] = 0.0f;
    for (int d = lane; d < DIM; d += 32) {
        float xv = xr[d];
#pragma unroll
        for (int e = 0; e < NEXP; ++e) acc[e] += xv * gw[e * DIM + d];
    }
#pragma unroll
    for (int e = 0; e < NEXP; ++e) {
        float v = acc[e];
        for (int off = 16; off; off >>= 1) v += __shfl_xor(v, off, 32);
        acc[e] = v;
    }
    if (lane == 0) {
        float m = acc[0]; int g = 0;
#pragma unroll
        for (int e = 1; e < NEXP; ++e) if (acc[e] > m) { m = acc[e]; g = e; }
        float ex[NEXP]; float den = 0.0f;
#pragma unroll
        for (int e = 0; e < NEXP; ++e) { ex[e] = expf(acc[e] - m); den += ex[e]; }
        float inv = 1.0f / den;
        expert_of[t] = g;
        w_of[t] = ex[g] * inv;
        atomicAdd(&sCnt[g], 1);
#pragma unroll
        for (int e = 0; e < NEXP; ++e) atomicAdd(&sProb[e], ex[e] * inv);
    }
    __syncthreads();
    if (tid < NEXP) {
        atomicAdd(&counts[tid], sCnt[tid]);
        atomicAdd(&probsum[tid], sProb[tid]);
    }
}

// ---------------- 2: build tile table (1 thread) ----------------
__global__ void assign_kernel(const int* __restrict__ counts,
                              int* base, int* cursor,
                              int* tileExpert, int* tileStart, int* nTiles)
{
    if (threadIdx.x == 0 && blockIdx.x == 0) {
        int s = 0, nt = 0;
        for (int e = 0; e < NEXP; ++e) {
            base[e] = s; cursor[e] = s;
            int c = counts[e];
            for (int off = 0; off < c; off += MT) {
                tileExpert[nt] = e; tileStart[nt] = s + off; ++nt;
            }
            s += c;
        }
        nTiles[0] = nt;
    }
}

// ---------------- 3: scatter tokens into per-expert lists ----------------
__global__ __launch_bounds__(256) void scatter_kernel(
    const int* __restrict__ expert_of, int* __restrict__ cursor,
    int* __restrict__ list)
{
    __shared__ int cnt[NEXP], bbase[NEXP], lcur[NEXP];
    const int tid = threadIdx.x;
    if (tid < NEXP) { cnt[tid] = 0; lcur[tid] = 0; }
    __syncthreads();
    const int t = blockIdx.x * 256 + tid;
    const int e = expert_of[t];
    atomicAdd(&cnt[e], 1);
    __syncthreads();
    if (tid < NEXP) bbase[tid] = atomicAdd(&cursor[tid], cnt[tid]);
    __syncthreads();
    int pos = bbase[e] + atomicAdd(&lcur[e], 1);
    list[pos] = t;
}

// ---------------- 3b: pre-swizzle weights fp32 -> bf16 WMMA B-fragments ----
// W1s fragment index: (e*W1_NT + nt)*W1_KS + ks  (ks contiguous -> stream 24KB/chunk)
// layout per fragment: 1024B, lane l's 8 packed dwords at +l*32
__global__ __launch_bounds__(256) void pack_w1_kernel(
    const float* __restrict__ W1, unsigned int* __restrict__ W1s)
{
    const int w = threadIdx.x >> 5, lane = threadIdx.x & 31;
    const int frag = blockIdx.x * 8 + w;
    const int ks = frag % W1_KS;
    const int tmp = frag / W1_KS;
    const int nt = tmp % W1_NT;
    const int e  = tmp / W1_NT;
    const int lq = lane & 15, half = lane >> 4;
    const float* wp = W1 + (size_t)e * DIM * DFF
                         + (size_t)(ks * 32 + half * 16) * DFF + nt * 16 + lq;
    unsigned int* dst = W1s + (size_t)frag * 256 + lane * 8;
#pragma unroll
    for (int j = 0; j < 8; ++j) {
        unsigned int lo = f2bf(wp[(size_t)(2 * j)     * DFF]);
        unsigned int hi = f2bf(wp[(size_t)(2 * j + 1) * DFF]);
        dst[j] = lo | (hi << 16);
    }
}

// W2s fragment index: (e*W2_KS + ks)*W2_NT + nt  (nt contiguous per k-step)
__global__ __launch_bounds__(256) void pack_w2_kernel(
    const float* __restrict__ W2, unsigned int* __restrict__ W2s)
{
    const int w = threadIdx.x >> 5, lane = threadIdx.x & 31;
    const int frag = blockIdx.x * 8 + w;
    const int nt = frag % W2_NT;
    const int tmp = frag / W2_NT;
    const int ks = tmp % W2_KS;
    const int e  = tmp / W2_KS;
    const int lq = lane & 15, half = lane >> 4;
    const float* wp = W2 + (size_t)e * DFF * DIM
                         + (size_t)(ks * 32 + half * 16) * DIM + nt * 16 + lq;
    unsigned int* dst = W2s + (size_t)frag * 256 + lane * 8;
#pragma unroll
    for (int j = 0; j < 8; ++j) {
        unsigned int lo = f2bf(wp[(size_t)(2 * j)     * DIM]);
        unsigned int hi = f2bf(wp[(size_t)(2 * j + 1) * DIM]);
        dst[j] = lo | (hi << 16);
    }
}

// ---------------- 4: fused expert FFN on WMMA ----------------
// block = 256 threads = 8 wave32; tile = 32 tokens x 768 out; h kept in LDS.
#define XSTR 776   // 768 + 8 bf16 pad (1552B row, conflict-free b128 reads)
#define HSTR 72    // 64 + 8 bf16 pad

template <bool SWZ>
__global__ __launch_bounds__(256) void ffn_kernel(
    const float* __restrict__ x,
    const float* __restrict__ W1, const float* __restrict__ b1,
    const float* __restrict__ W2, const float* __restrict__ b2,
    const unsigned int* __restrict__ W1s, const unsigned int* __restrict__ W2s,
    const int* __restrict__ tileExpert, const int* __restrict__ tileStart,
    const int* __restrict__ nTiles,
    const int* __restrict__ base, const int* __restrict__ counts,
    const int* __restrict__ list, const float* __restrict__ w_of,
    float* __restrict__ out)
{
    __shared__ __align__(16) unsigned short xs[MT * XSTR];  // 49.6 KB
    __shared__ __align__(16) unsigned short hs[MT * HSTR];  //  4.6 KB

    const int bid = blockIdx.x;
    if (bid >= nTiles[0]) return;                 // block-uniform exit

    const int e      = tileExpert[bid];
    const int tstart = tileStart[bid];
    const int eEnd   = base[e] + counts[e];
    const int valid  = (eEnd - tstart) < MT ? (eEnd - tstart) : MT;

    const int tid  = threadIdx.x;
    const int lane = tid & 31;
    const int w    = tid >> 5;
    const int halfSel = lane >> 4;                // 0 | 1
    const int lq      = lane & 15;

    const float* __restrict__ W1e = W1 + (size_t)e * DIM * DFF;
    const float* __restrict__ W2e = W2 + (size_t)e * DFF * DIM;
    const float* __restrict__ b1e = b1 + (size_t)e * DFF;
    const float* __restrict__ b2e = b2 + (size_t)e * DIM;

    // ---- stage x tile (gathered token rows) as bf16 into LDS ----
    for (int idx = tid; idx < MT * (DIM / 4); idx += 256) {
        int r  = idx / (DIM / 4);
        int c4 = idx % (DIM / 4);
        int tok = list[tstart + (r < valid ? r : 0)];
        float4 xv = ((const float4*)(x + (size_t)tok * DIM))[c4];
        int o = r * XSTR + c4 * 4;
        xs[o + 0] = f2bf(xv.x); xs[o + 1] = f2bf(xv.y);
        xs[o + 2] = f2bf(xv.z); xs[o + 3] = f2bf(xv.w);
    }
    __syncthreads();

    const f32x8 zf = {0.f, 0.f, 0.f, 0.f, 0.f, 0.f, 0.f, 0.f};
    f32x8 acc2[12];                               // 2 M-tiles x 6 N-tiles
#pragma unroll
    for (int i = 0; i < 12; ++i) acc2[i] = zf;

    const int m1  = w & 1;                        // GEMM1 role of this wave
    const int n1  = w >> 1;
    const int nb0 = w * 96;                       // GEMM2 N base (6 tiles)

    for (int ch = 0; ch < DFF / CHUNK; ++ch) {
        const int cb = ch * CHUNK;

        // warm the next chunk's fragments in L2 (cold-pass helper)
        if (SWZ && (ch + 1 < DFF / CHUNK)) {
            const char* p1 = (const char*)(W1s +
                ((size_t)(e * W1_NT + (ch + 1) * 4 + n1) * W1_KS) * 256);
            const char* p2 = (const char*)(W2s +
                ((size_t)(e * W2_KS + (ch + 1) * 2) * W2_NT + w * 6) * 256);
#pragma unroll
            for (int i = 0; i < 6; ++i)           // 24KB of W1s per wave
                __builtin_prefetch(p1 + lane * 128 + i * 4096, 0, 3);
#pragma unroll
            for (int i = 0; i < 3; ++i)           // 12KB of W2s per wave
                __builtin_prefetch(p2 + lane * 128 + i * 4096, 0, 3);
        }

        // ---------- GEMM1: h(32x64) = x(32x768) * W1[:, cb:cb+64] ----------
        f32x8 acc1 = zf;
        const int colB1 = cb + n1 * 16 + lq;      // W1 column (fallback path)
        const int nt1   = ch * 4 + n1;            // W1s N-tile (swizzled path)
        const unsigned int* fragBase1 =
            W1s + ((size_t)(e * W1_NT + nt1) * W1_KS) * 256 + lane * 8;

        Frag Bcur;
        if (SWZ) loadFragG(fragBase1, Bcur);
        for (int ks = 0; ks < DIM / 32; ++ks) {
            Frag A;
            {
                int row   = m1 * 16 + lq;
                int kbase = ks * 32 + halfSel * 16;
                uint4 a0 = *(const uint4*)&xs[row * XSTR + kbase];
                uint4 a1 = *(const uint4*)&xs[row * XSTR + kbase + 8];
                A.u[0] = a0.x; A.u[1] = a0.y; A.u[2] = a0.z; A.u[3] = a0.w;
                A.u[4] = a1.x; A.u[5] = a1.y; A.u[6] = a1.z; A.u[7] = a1.w;
            }
            if (SWZ) {
                Frag Bnxt;                        // prefetch next fragment
                int ksn = (ks + 1 < DIM / 32) ? ks + 1 : ks;
                loadFragG(fragBase1 + (size_t)ksn * 256, Bnxt);
                acc1 = __builtin_amdgcn_wmma_f32_16x16x32_bf16(
                           false, A.v, false, Bcur.v, (short)0, acc1, false, false);
                Bcur = Bnxt;
            } else {
                Frag B;
                const float* wp = W1e + (size_t)(ks * 32 + halfSel * 16) * DFF + colB1;
#pragma unroll
                for (int j = 0; j < 8; ++j) {
                    unsigned int lo = f2bf(wp[(size_t)(2 * j)     * DFF]);
                    unsigned int hi = f2bf(wp[(size_t)(2 * j + 1) * DFF]);
                    B.u[j] = lo | (hi << 16);
                }
                acc1 = __builtin_amdgcn_wmma_f32_16x16x32_bf16(
                           false, A.v, false, B.v, (short)0, acc1, false, false);
            }
        }
        // bias + exact gelu, write bf16 h tile to LDS
        {
            const float bias1 = b1e[cb + n1 * 16 + lq];
            const int hcol = n1 * 16 + lq;
#pragma unroll
            for (int v = 0; v < 8; ++v) {
                float hv = acc1[v] + bias1;
                hv = 0.5f * hv * (1.0f + erff(hv * 0.70710678118654752f));
                int hrow = m1 * 16 + halfSel * 8 + v;
                hs[hrow * HSTR + hcol] = f2bf(hv);
            }
        }
        __syncthreads();

        // ---------- GEMM2: y(32x768) += h(32x64) * W2[cb:cb+64, :] ----------
#pragma unroll
        for (int k2 = 0; k2 < 2; ++k2) {
            Frag A0, A1;
            {
                int kbase = k2 * 32 + halfSel * 16;
                uint4 a0 = *(const uint4*)&hs[lq * HSTR + kbase];
                uint4 a1 = *(const uint4*)&hs[lq * HSTR + kbase + 8];
                A0.u[0] = a0.x; A0.u[1] = a0.y; A0.u[2] = a0.z; A0.u[3] = a0.w;
                A0.u[4] = a1.x; A0.u[5] = a1.y; A0.u[6] = a1.z; A0.u[7] = a1.w;
                uint4 b0 = *(const uint4*)&hs[(16 + lq) * HSTR + kbase];
                uint4 b1v = *(const uint4*)&hs[(16 + lq) * HSTR + kbase + 8];
                A1.u[0] = b0.x; A1.u[1] = b0.y; A1.u[2] = b0.z; A1.u[3] = b0.w;
                A1.u[4] = b1v.x; A1.u[5] = b1v.y; A1.u[6] = b1v.z; A1.u[7] = b1v.w;
            }
            const int ks2 = ch * 2 + k2;
            const unsigned int* fragBase2 =
                W2s + ((size_t)(e * W2_KS + ks2) * W2_NT + w * 6) * 256 + lane * 8;

            if (SWZ) {
                Frag Bc; loadFragG(fragBase2, Bc);
#pragma unroll
                for (int n = 0; n < 6; ++n) {
                    Frag Bn2;                     // prefetch next fragment
                    int nn = (n + 1 < 6) ? n + 1 : n;
                    loadFragG(fragBase2 + (size_t)nn * 256, Bn2);
                    acc2[n * 2 + 0] = __builtin_amdgcn_wmma_f32_16x16x32_bf16(
                        false, A0.v, false, Bc.v, (short)0, acc2[n * 2 + 0], false, false);
                    acc2[n * 2 + 1] = __builtin_amdgcn_wmma_f32_16x16x32_bf16(
                        false, A1.v, false, Bc.v, (short)0, acc2[n * 2 + 1], false, false);
                    Bc = Bn2;
                }
            } else {
#pragma unroll
                for (int n = 0; n < 6; ++n) {
                    Frag B;
                    const int col = nb0 + n * 16 + lq;
                    const float* wp = W2e + (size_t)(cb + k2 * 32 + halfSel * 16) * DIM + col;
#pragma unroll
                    for (int j = 0; j < 8; ++j) {
                        unsigned int lo = f2bf(wp[(size_t)(2 * j)     * DIM]);
                        unsigned int hi = f2bf(wp[(size_t)(2 * j + 1) * DIM]);
                        B.u[j] = lo | (hi << 16);
                    }
                    acc2[n * 2 + 0] = __builtin_amdgcn_wmma_f32_16x16x32_bf16(
                        false, A0.v, false, B.v, (short)0, acc2[n * 2 + 0], false, false);
                    acc2[n * 2 + 1] = __builtin_amdgcn_wmma_f32_16x16x32_bf16(
                        false, A1.v, false, B.v, (short)0, acc2[n * 2 + 1], false, false);
                }
            }
        }
        __syncthreads();    // protect hs before next chunk overwrites it
    }

    // ---- epilogue: +b2, scale by selected prob, scatter to out ----
#pragma unroll
    for (int mt = 0; mt < 2; ++mt) {
        float wsel[8]; int tokid[8]; bool ok[8];
#pragma unroll
        for (int v = 0; v < 8; ++v) {
            int ra = mt * 16 + halfSel * 8 + v;
            ok[v] = (ra < valid);
            int tok = list[tstart + (ok[v] ? ra : 0)];
            tokid[v] = tok;
            wsel[v] = w_of[tok];
        }
#pragma unroll
        for (int n = 0; n < 6; ++n) {
            const int col = nb0 + n * 16 + lq;
            const float bias2 = b2e[col];
            f32x8 a = acc2[n * 2 + mt];
#pragma unroll
            for (int v = 0; v < 8; ++v) {
                if (ok[v])
                    out[(size_t)tokid[v] * DIM + col] = wsel[v] * (a[v] + bias2);
            }
        }
    }
}

// ---------------- 5: balance loss + gate_load ----------------
__global__ void finalize_kernel(const int* __restrict__ counts,
                                const float* __restrict__ probsum,
                                float* __restrict__ out_tail)
{
    if (threadIdx.x == 0 && blockIdx.x == 0) {
        const float invT = 1.0f / (float)NTOK;
        float loss = 0.0f;
        for (int e = 0; e < NEXP; ++e)
            loss += (probsum[e] * invT) * ((float)counts[e] * invT);
        out_tail[0] = (float)NEXP * loss;
        for (int e = 0; e < NEXP; ++e)
            out_tail[1 + e] = (float)counts[e];
    }
}

// ---------------- host launcher ----------------
extern "C" void kernel_launch(void* const* d_in, const int* in_sizes, int n_in,
                              void* d_out, int out_size, void* d_ws, size_t ws_size,
                              hipStream_t stream)
{
    (void)in_sizes; (void)n_in; (void)out_size;
    const float* x  = (const float*)d_in[0];
    // d_in[1] = attention_mask (all ones; unused by the reference math)
    const float* gw = (const float*)d_in[2];
    const float* W1 = (const float*)d_in[3];
    const float* b1 = (const float*)d_in[4];
    const float* W2 = (const float*)d_in[5];
    const float* b2 = (const float*)d_in[6];
    float* out = (float*)d_out;

    // workspace carve-up (control block first, then optional bf16 weights)
    char* p = (char*)d_ws;
    int*   expert_of = (int*)p;            p += (size_t)NTOK * 4;
    float* w_of      = (float*)p;          p += (size_t)NTOK * 4;
    int*   counts    = (int*)p;            p += NEXP * 4;
    float* probsum   = (float*)p;          p += NEXP * 4;
    int*   base      = (int*)p;            p += NEXP * 4;
    int*   cursor    = (int*)p;            p += NEXP * 4;
    int*   nTiles    = (int*)p;            p += 16;
    int*   tileExpert= (int*)p;            p += (size_t)MAXTILES * 4;
    int*   tileStart = (int*)p;            p += (size_t)MAXTILES * 4;
    int*   list      = (int*)p;            p += (size_t)NTOK * 4;
    // align weight buffers to 1KB
    size_t ctrl = (size_t)(p - (char*)d_ws);
    ctrl = (ctrl + 1023) & ~(size_t)1023;
    unsigned int* W1s = (unsigned int*)((char*)d_ws + ctrl);
    unsigned int* W2s = (unsigned int*)((char*)d_ws + ctrl + (size_t)W1_FRAGS * 1024);
    const size_t needed = ctrl + ((size_t)W1_FRAGS + (size_t)W2_FRAGS) * 1024;
    const bool useSwz = (ws_size >= needed);

    init_kernel<<<1, 64, 0, stream>>>(counts, probsum);
    gate_kernel<<<NTOK / 8, 256, 0, stream>>>(x, gw, expert_of, w_of, counts, probsum);
    assign_kernel<<<1, 32, 0, stream>>>(counts, base, cursor, tileExpert, tileStart, nTiles);
    scatter_kernel<<<NTOK / 256, 256, 0, stream>>>(expert_of, cursor, list);
    finalize_kernel<<<1, 32, 0, stream>>>(counts, probsum, out + (size_t)OUT_ELEMS);

    if (useSwz) {
        pack_w1_kernel<<<W1_FRAGS / 8, 256, 0, stream>>>(W1, W1s);
        pack_w2_kernel<<<W2_FRAGS / 8, 256, 0, stream>>>(W2, W2s);
        ffn_kernel<true><<<MAXTILES, 256, 0, stream>>>(
            x, W1, b1, W2, b2, W1s, W2s,
            tileExpert, tileStart, nTiles, base, counts, list, w_of, out);
    } else {
        ffn_kernel<false><<<MAXTILES, 256, 0, stream>>>(
            x, W1, b1, W2, b2, W1s, W2s,
            tileExpert, tileStart, nTiles, base, counts, list, w_of, out);
    }
}
